// OpenAIMoeDecoderLayer_34136400068636
// MI455X (gfx1250) — matrix operand order repaired and checked by reading, not compiled, not run on hardware
//
#include <hip/hip_runtime.h>
#include <hip/hip_bf16.h>
#include <string.h>

// ---------------------------------------------------------------------------
// Problem constants
// ---------------------------------------------------------------------------
#define S_LEN   2048
#define HIDDEN  2880
#define HQ      64
#define HKV     8
#define HDIM    64
#define NQKV    ((HQ + 2*HKV) * HDIM)   // 5120
#define WINDOW  127
#define THETA   150000.0f

typedef __attribute__((ext_vector_type(16))) __bf16        v16bf;
typedef __attribute__((ext_vector_type(8)))  float         v8f;
typedef __attribute__((ext_vector_type(4)))  unsigned int  v4u;
typedef __attribute__((ext_vector_type(8)))  int           v8i;
typedef __attribute__((ext_vector_type(4)))  int           v4i;

// TDM builtin: 5-arg on ROCm clang-22, 6-arg on upstream clang-23 (therock).
#if defined(__has_builtin)
#if __has_builtin(__builtin_amdgcn_tensor_load_to_lds)
#define USE_TDM 1
#endif
#endif

union FragB {
    uint4  u[2];
    v16bf  v;
};

static __device__ __forceinline__ unsigned short f2bf(float x) {
    unsigned int u = __float_as_uint(x);
    unsigned int r = u + 0x7FFFu + ((u >> 16) & 1u);   // round-to-nearest-even
    return (unsigned short)(r >> 16);
}

static __device__ __forceinline__ v8f wmma_bf16(v16bf a, v16bf b, v8f c) {
    // 8 args: (neg_a, A, neg_b, B, c_mod, C, reuse_a, reuse_b)
    return __builtin_amdgcn_wmma_f32_16x16x32_bf16(false, a, false, b,
                                                   (short)0, c, false, false);
}

#ifdef USE_TDM
// ---------------------------------------------------------------------------
// Tensor Data Mover: 2-D tile load (bf16) global -> LDS.
//   tile: tile_h rows x 32 elements (64 B/row), tensor row stride in elements.
//   LDS padding via D#: every 16 DWORDs insert 4 DWORDs -> 80 B LDS row
//   stride == LDA shorts.  Rows >= avail_rows read as zero (OOB fill).
// Descriptor bitfields per CDNA5 ISA sec 8.3-8.6.
// ---------------------------------------------------------------------------
static __device__ __forceinline__ void tdm_load_tile_2d(
    void* lds_dst, const void* gsrc,
    unsigned tile_h, unsigned avail_rows, unsigned row_stride_elems)
{
    const unsigned long long ga = (unsigned long long)gsrc;
    v4u g0;
    g0.x = 1u;                                                 // count=1, user mode
    g0.y = (unsigned)(unsigned long long)lds_dst;              // lds byte offset
    g0.z = (unsigned)ga;                                       // global_addr[31:0]
    g0.w = (unsigned)((ga >> 32) & 0x01FFFFFFu) | (2u << 30);  // addr[56:32], type=2

    const unsigned tw = 32;                                    // tile/tensor dim0
    const unsigned flags = (1u << 16)    // data_size = 2 bytes
                         | (1u << 20)    // pad_enable
                         | (3u << 22)    // pad_interval: 16 DWORDs (64 B)
                         | (3u << 25);   // pad_amount:   4 DWORDs (16 B)
    v8i g1;
    g1[0] = (int)flags;
    g1[1] = (int)((tw & 0xFFFFu) << 16);                               // dim0 lo
    g1[2] = (int)((tw >> 16) | ((avail_rows & 0xFFFFu) << 16));        // dim0 hi | dim1 lo
    g1[3] = (int)((avail_rows >> 16) | (tw << 16));                    // dim1 hi | tile_dim0
    g1[4] = (int)(tile_h & 0xFFFFu);                                   // tile_dim1, tile_dim2=0
    g1[5] = (int)row_stride_elems;                                     // dim0_stride[31:0]
    g1[6] = 0;                                                         // stride hi, dim1_stride lo
    g1[7] = 0;

    v4i g2; g2[0] = 1; g2[1] = 1; g2[2] = 0; g2[3] = 0;   // dim2=1, dim3=1, tile_dim3=0
    v4i g3; g3[0] = 0; g3[1] = (int)(1u << 16); g3[2] = 0; g3[3] = 0;  // dim4=1, tile_dim4=0

#if defined(__clang_major__) && (__clang_major__ < 23)
    __builtin_amdgcn_tensor_load_to_lds(g0, g1, g2, g3, 0);
#else
    v8i gx = {};   // clang-23 6-arg form: extra int32x8 operand (zero)
    __builtin_amdgcn_tensor_load_to_lds(g0, g1, g2, g3, gx, 0);
#endif
}
#endif

// ---------------------------------------------------------------------------
// Kernel 1: fp32 -> bf16 conversion (grid-stride)
// ---------------------------------------------------------------------------
__global__ void k_f32_to_bf16(const float* __restrict__ in,
                              unsigned short* __restrict__ out, size_t n) {
    size_t i = (size_t)blockIdx.x * blockDim.x + threadIdx.x;
    size_t stride = (size_t)gridDim.x * blockDim.x;
    for (; i < n; i += stride) out[i] = f2bf(in[i]);
}

// ---------------------------------------------------------------------------
// Kernel 2/5: C(MxN, fp32) = A(MxK, bf16) * W(NxK, bf16)^T
// Block tile 128x256, K-step 32, 256 threads = 8 waves (2x4), wave tile 64x64.
// Double-buffered LDS; staging via TDM (wave 0 issues descriptor DMAs) when
// available, else cooperative vector loads.
// ---------------------------------------------------------------------------
#define GBM 128
#define GBN 256
#define LDA 40   // padded row stride in shorts (80 B = 5*16, keeps b128 aligned)

__global__ __launch_bounds__(256) void k_gemm_xwt(
    const unsigned short* __restrict__ A,
    const unsigned short* __restrict__ W,
    float* __restrict__ C,
    int M, int N, int K)
{
    __shared__ __align__(16) unsigned short As[2][GBM][LDA];  // 2 x 10240 B
    __shared__ __align__(16) unsigned short Bs[2][GBN][LDA];  // 2 x 20480 B

    const int t      = threadIdx.x;
    const int wave   = t >> 5;
    const int lane   = t & 31;
    const int lane16 = lane & 15;
    const int hi     = lane >> 4;
    const int wm     = (wave >> 2) * 64;   // wave M offset in tile (0,64)
    const int wn     = (wave & 3) * 64;    // wave N offset in tile (0..192)
    const int m0     = blockIdx.y * GBM;
    const int n0     = blockIdx.x * GBN;

    v8f zero = {};
    v8f acc[4][4];
    for (int mt = 0; mt < 4; ++mt)
        for (int nt = 0; nt < 4; ++nt) acc[mt][nt] = zero;

    const int nk = K >> 5;

#ifdef USE_TDM
    // ---- TDM staging: one wave issues two descriptor DMAs per K-step ----
    const bool issuer = (wave == 0);
    int avail = N - n0; if (avail > GBN) avail = GBN; if (avail < 0) avail = 0;

    if (issuer) {
        tdm_load_tile_2d(&As[0][0][0], A + (size_t)m0 * K, GBM, GBM, (unsigned)K);
        tdm_load_tile_2d(&Bs[0][0][0], W + (size_t)n0 * K, GBN, (unsigned)avail, (unsigned)K);
        __builtin_amdgcn_s_wait_tensorcnt(0);
    }
    __syncthreads();

    for (int kk = 0; kk < nk; ++kk) {
        const int cur = kk & 1;
        if (issuer && (kk + 1) < nk) {
            const int koff = (kk + 1) << 5;
            tdm_load_tile_2d(&As[cur ^ 1][0][0], A + (size_t)m0 * K + koff,
                             GBM, GBM, (unsigned)K);
            tdm_load_tile_2d(&Bs[cur ^ 1][0][0], W + (size_t)n0 * K + koff,
                             GBN, (unsigned)avail, (unsigned)K);
        }

        FragB af[4], bf[4];
        for (int mt = 0; mt < 4; ++mt) {
            const int r = wm + mt * 16 + lane16;
            af[mt].u[0] = *(const uint4*)&As[cur][r][hi * 8];
            af[mt].u[1] = *(const uint4*)&As[cur][r][16 + hi * 8];
        }
        for (int nt = 0; nt < 4; ++nt) {
            const int r = wn + nt * 16 + lane16;
            bf[nt].u[0] = *(const uint4*)&Bs[cur][r][hi * 8];
            bf[nt].u[1] = *(const uint4*)&Bs[cur][r][16 + hi * 8];
        }
        for (int mt = 0; mt < 4; ++mt)
            for (int nt = 0; nt < 4; ++nt)
                acc[mt][nt] = wmma_bf16(af[mt].v, bf[nt].v, acc[mt][nt]);

        if (issuer) __builtin_amdgcn_s_wait_tensorcnt(0);
        __syncthreads();
    }
#else
    // ---- cooperative staging fallback ----
    const int alr = t >> 1;
    const int alc = (t & 1) * 16;
    const int blr = t;
    const bool bvalid = (n0 + blr) < N;
    const unsigned short* aptr = A + (size_t)(m0 + alr) * K + alc;
    const unsigned short* bptr = W + (size_t)(n0 + blr) * K;

    {
        const uint4* sa = (const uint4*)aptr;
        *(uint4*)&As[0][alr][alc]     = sa[0];
        *(uint4*)&As[0][alr][alc + 8] = sa[1];
        uint4 b0 = {0,0,0,0}, b1 = b0, b2 = b0, b3 = b0;
        if (bvalid) {
            const uint4* sb = (const uint4*)bptr;
            b0 = sb[0]; b1 = sb[1]; b2 = sb[2]; b3 = sb[3];
        }
        *(uint4*)&Bs[0][blr][0]  = b0;
        *(uint4*)&Bs[0][blr][8]  = b1;
        *(uint4*)&Bs[0][blr][16] = b2;
        *(uint4*)&Bs[0][blr][24] = b3;
    }
    __syncthreads();

    for (int kk = 0; kk < nk; ++kk) {
        const int cur = kk & 1;

        uint4 pa0, pa1, pb0, pb1, pb2, pb3;
        const bool havenext = (kk + 1) < nk;
        if (havenext) {
            const int koff = (kk + 1) << 5;
            const uint4* sa = (const uint4*)(aptr + koff);
            pa0 = sa[0]; pa1 = sa[1];
            pb0 = (uint4){0,0,0,0}; pb1 = pb0; pb2 = pb0; pb3 = pb0;
            if (bvalid) {
                const uint4* sb = (const uint4*)(bptr + koff);
                pb0 = sb[0]; pb1 = sb[1]; pb2 = sb[2]; pb3 = sb[3];
            }
        }

        FragB af[4], bf[4];
        for (int mt = 0; mt < 4; ++mt) {
            const int r = wm + mt * 16 + lane16;
            af[mt].u[0] = *(const uint4*)&As[cur][r][hi * 8];
            af[mt].u[1] = *(const uint4*)&As[cur][r][16 + hi * 8];
        }
        for (int nt = 0; nt < 4; ++nt) {
            const int r = wn + nt * 16 + lane16;
            bf[nt].u[0] = *(const uint4*)&Bs[cur][r][hi * 8];
            bf[nt].u[1] = *(const uint4*)&Bs[cur][r][16 + hi * 8];
        }
        for (int mt = 0; mt < 4; ++mt)
            for (int nt = 0; nt < 4; ++nt)
                acc[mt][nt] = wmma_bf16(af[mt].v, bf[nt].v, acc[mt][nt]);

        if (havenext) {
            const int nxt = cur ^ 1;
            *(uint4*)&As[nxt][alr][alc]     = pa0;
            *(uint4*)&As[nxt][alr][alc + 8] = pa1;
            *(uint4*)&Bs[nxt][blr][0]  = pb0;
            *(uint4*)&Bs[nxt][blr][8]  = pb1;
            *(uint4*)&Bs[nxt][blr][16] = pb2;
            *(uint4*)&Bs[nxt][blr][24] = pb3;
        }
        __syncthreads();
    }
#endif

    // ---- store C (C-layout: VGPR i -> row hi*8+i, col = lane16) ----
    for (int mt = 0; mt < 4; ++mt)
        for (int nt = 0; nt < 4; ++nt)
            for (int i = 0; i < 8; ++i) {
                const int row = m0 + wm + mt * 16 + hi * 8 + i;
                const int col = n0 + wn + nt * 16 + lane16;
                if (col < N) C[(size_t)row * N + col] = acc[mt][nt][i];
            }
}

// ---------------------------------------------------------------------------
// Kernel 3: RoPE on Q/K heads + head split + bf16 convert (one block per seq)
// ---------------------------------------------------------------------------
__global__ void k_rope_split(const float* __restrict__ qkv,
                             const int* __restrict__ positions,
                             unsigned short* __restrict__ qb,
                             unsigned short* __restrict__ kb,
                             unsigned short* __restrict__ vb)
{
    const int   s   = blockIdx.x;
    const float pos = (float)positions[s];
    const float* row = qkv + (size_t)s * NQKV;
    const float lth = logf(THETA) * (1.0f / 32.0f);

    // 72 rope heads (64 q + 8 k) x 32 freq pairs
    for (int idx = threadIdx.x; idx < 72 * 32; idx += blockDim.x) {
        const int head = idx >> 5;
        const int j    = idx & 31;
        const float x1 = row[head * 64 + j];
        const float x2 = row[head * 64 + j + 32];
        const float fr = pos * expf(-(float)j * lth);
        float sn, cs;
        sincosf(fr, &sn, &cs);
        const float o1 = x1 * cs - x2 * sn;
        const float o2 = x2 * cs + x1 * sn;
        unsigned short* dst = (head < HQ)
            ? qb + ((size_t)s * HQ + head) * HDIM
            : kb + ((size_t)s * HKV + (head - HQ)) * HDIM;
        dst[j]      = f2bf(o1);
        dst[j + 32] = f2bf(o2);
    }
    // V heads: straight convert
    for (int idx = threadIdx.x; idx < HKV * HDIM; idx += blockDim.x)
        vb[(size_t)s * (HKV * HDIM) + idx] = f2bf(row[(HQ + HKV) * HDIM + idx]);
}

// ---------------------------------------------------------------------------
// Kernel 4: sliding-window flash attention with sinks.
// One wave (32 threads) per (head, 16-query block). 5 chunks of 32 keys.
// ---------------------------------------------------------------------------
__global__ __launch_bounds__(32) void k_attn(
    const unsigned short* __restrict__ q,
    const unsigned short* __restrict__ k,
    const unsigned short* __restrict__ v,
    const float* __restrict__ sinks,
    unsigned short* __restrict__ out)
{
    __shared__ __align__(16) unsigned short Qs[16][64];   // [qrow][d]
    __shared__ __align__(16) unsigned short Ks[32][64];   // [key][d]
    __shared__ __align__(16) unsigned short Vt[64][48];   // [d][key]  (stride 96B)
    __shared__ __align__(16) unsigned short Ps[16][48];   // [qrow][key]

    const int h      = blockIdx.x;
    const int m0     = blockIdx.y * 16;
    const int kvh    = h >> 3;               // HQ/HKV = 8
    const int lane   = threadIdx.x;
    const int lane16 = lane & 15;
    const int hi     = lane >> 4;
    const float sm_scale = 0.125f;           // 1/sqrt(64)

    // ---- load Q tile (16 x 64 bf16) ----
    {
        const int r = lane >> 1, c = (lane & 1) * 32;
        const uint4* src = (const uint4*)(q + ((size_t)(m0 + r) * HQ + h) * HDIM + c);
        uint4* dst = (uint4*)&Qs[r][c];
        dst[0] = src[0]; dst[1] = src[1]; dst[2] = src[2]; dst[3] = src[3];
    }
    __syncthreads();

    FragB aq[2];
    for (int kc = 0; kc < 2; ++kc) {
        aq[kc].u[0] = *(const uint4*)&Qs[lane16][kc * 32 + hi * 8];
        aq[kc].u[1] = *(const uint4*)&Qs[lane16][kc * 32 + 16 + hi * 8];
    }

    v8f zero = {};
    v8f acco[4];
    for (int i = 0; i < 4; ++i) acco[i] = zero;
    float mrow[8], lrow[8];
    const float lsink = logf(sinks[h]);
    for (int i = 0; i < 8; ++i) { mrow[i] = lsink; lrow[i] = 1.0f; }

    for (int c = 0; c < 5; ++c) {
        const int kb = m0 - 127 + 32 * c;    // chunk covers keys kb..kb+31
        __syncthreads();                     // protect Ks/Vt/Ps reuse

        // ---- load K chunk (zero-fill out-of-range keys) ----
        for (int it = 0; it < 2; ++it) {
            const int task = lane + it * 32;
            const int r    = task >> 1;
            const int cc   = (task & 1) * 32;
            const int key  = kb + r;
            uint4 d0 = {0,0,0,0}, d1 = d0, d2 = d0, d3 = d0;
            if (key >= 0 && key < S_LEN) {
                const uint4* src = (const uint4*)(k + ((size_t)key * HKV + kvh) * HDIM + cc);
                d0 = src[0]; d1 = src[1]; d2 = src[2]; d3 = src[3];
            }
            uint4* dst = (uint4*)&Ks[r][cc];
            dst[0] = d0; dst[1] = d1; dst[2] = d2; dst[3] = d3;
        }
        // ---- load V chunk transposed: Vt[d][key] ----
        {
            const int key = kb + lane;
            union { uint4 q4[8]; unsigned short s16[64]; } vbuf;
            for (int j = 0; j < 8; ++j) vbuf.q4[j] = (uint4){0,0,0,0};
            if (key >= 0 && key < S_LEN) {
                const uint4* src = (const uint4*)(v + ((size_t)key * HKV + kvh) * HDIM);
                for (int j = 0; j < 8; ++j) vbuf.q4[j] = src[j];
            }
            for (int d = 0; d < 64; ++d) Vt[d][lane] = vbuf.s16[d];
        }
        __syncthreads();

        // ---- logits: Q(16x64) x K^T(64x32) via 2 N-tiles x 2 K-chunks ----
        v8f lo[2];
        for (int nt = 0; nt < 2; ++nt) {
            v8f cacc = zero;
            for (int kc = 0; kc < 2; ++kc) {
                FragB bk;
                const int r = nt * 16 + lane16;
                bk.u[0] = *(const uint4*)&Ks[r][kc * 32 + hi * 8];
                bk.u[1] = *(const uint4*)&Ks[r][kc * 32 + 16 + hi * 8];
                cacc = wmma_bf16(aq[kc].v, bk.v, cacc);
            }
            lo[nt] = cacc;
        }

        // ---- online softmax (rows live in registers, cols across 16 lanes) --
        float pt[2][8], tmax[8];
        for (int i = 0; i < 8; ++i) {
            const int row = m0 + hi * 8 + i;
            float best = -INFINITY;
            for (int nt = 0; nt < 2; ++nt) {
                const int col = kb + nt * 16 + lane16;
                const bool valid = (col >= 0) && (col <= row) && (col >= row - WINDOW);
                const float lg = valid ? lo[nt][i] * sm_scale : -INFINITY;
                pt[nt][i] = lg;
                best = fmaxf(best, lg);
            }
            tmax[i] = best;
        }
        for (int i = 0; i < 8; ++i)
            for (int sft = 1; sft < 16; sft <<= 1)
                tmax[i] = fmaxf(tmax[i], __shfl_xor(tmax[i], sft, 32));

        float resc[8];
        for (int i = 0; i < 8; ++i) {
            const float nm = fmaxf(mrow[i], tmax[i]);
            resc[i] = expf(mrow[i] - nm);
            mrow[i] = nm;
        }
        float rsum[8];
        for (int i = 0; i < 8; ++i) {
            float sacc = 0.0f;
            for (int nt = 0; nt < 2; ++nt) {
                const float p = expf(pt[nt][i] - mrow[i]);  // masked -> exp(-inf)=0
                pt[nt][i] = p;
                sacc += p;
            }
            rsum[i] = sacc;
        }
        for (int i = 0; i < 8; ++i)
            for (int sft = 1; sft < 16; sft <<= 1)
                rsum[i] += __shfl_xor(rsum[i], sft, 32);
        for (int i = 0; i < 8; ++i) lrow[i] = lrow[i] * resc[i] + rsum[i];
        for (int nt = 0; nt < 4; ++nt)
            for (int i = 0; i < 8; ++i) acco[nt][i] *= resc[i];

        // ---- P (C-layout) -> LDS -> A-fragment layout ----
        for (int nt = 0; nt < 2; ++nt)
            for (int i = 0; i < 8; ++i)
                Ps[hi * 8 + i][nt * 16 + lane16] = f2bf(pt[nt][i]);
        __syncthreads();

        FragB ap;
        ap.u[0] = *(const uint4*)&Ps[lane16][hi * 8];
        ap.u[1] = *(const uint4*)&Ps[lane16][16 + hi * 8];

        // ---- O += P(16x32) x V(32x64), 4 N-tiles of head-dim ----
        for (int nt = 0; nt < 4; ++nt) {
            FragB bv;
            const int r = nt * 16 + lane16;
            bv.u[0] = *(const uint4*)&Vt[r][hi * 8];
            bv.u[1] = *(const uint4*)&Vt[r][16 + hi * 8];
            acco[nt] = wmma_bf16(ap.v, bv.v, acco[nt]);
        }
    }

    // ---- epilogue: normalize and store bf16 [s][h*64+d] ----
    for (int nt = 0; nt < 4; ++nt)
        for (int i = 0; i < 8; ++i) {
            const int row = m0 + hi * 8 + i;
            const int d   = nt * 16 + lane16;
            out[(size_t)row * (HQ * HDIM) + h * HDIM + d] =
                f2bf(acco[nt][i] / lrow[i]);
        }
}

// ---------------------------------------------------------------------------
// Host: workspace layout + launch sequence
// ---------------------------------------------------------------------------
extern "C" void kernel_launch(void* const* d_in, const int* in_sizes, int n_in,
                              void* d_out, int out_size, void* d_ws, size_t ws_size,
                              hipStream_t stream) {
    const float* hidden    = (const float*)d_in[0];  // S x HIDDEN
    const float* qkv_w     = (const float*)d_in[1];  // NQKV x HIDDEN
    const float* o_w       = (const float*)d_in[2];  // HIDDEN x HQ*D
    const float* sinks     = (const float*)d_in[3];  // HQ
    const int*   positions = (const int*)d_in[4];    // S
    float*       outp      = (float*)d_out;

    char* ws = (char*)d_ws;
    size_t off = 0;
    unsigned short* hidden_b = (unsigned short*)(ws + off); off += (size_t)S_LEN * HIDDEN * 2;      // 11.8 MB
    unsigned short* qkvw_b   = (unsigned short*)(ws + off); off += (size_t)NQKV * HIDDEN * 2;       // 29.5 MB
    unsigned short* ow_b     = (unsigned short*)(ws + off); off += (size_t)HIDDEN * HQ * HDIM * 2;  // 23.6 MB
    float*          qkv_f    = (float*)(ws + off);          off += (size_t)S_LEN * NQKV * 4;        // 41.9 MB
    unsigned short* q_b      = (unsigned short*)(ws + off); off += (size_t)S_LEN * HQ * HDIM * 2;   // 16.8 MB
    unsigned short* k_b      = (unsigned short*)(ws + off); off += (size_t)S_LEN * HKV * HDIM * 2;  //  2.1 MB
    unsigned short* v_b      = (unsigned short*)(ws + off); off += (size_t)S_LEN * HKV * HDIM * 2;  //  2.1 MB
    unsigned short* attn_b   = (unsigned short*)(ws + off); off += (size_t)S_LEN * HQ * HDIM * 2;   // 16.8 MB

    // 1) fp32 -> bf16 conversions
    k_f32_to_bf16<<<4096, 256, 0, stream>>>(hidden, hidden_b, (size_t)S_LEN * HIDDEN);
    k_f32_to_bf16<<<4096, 256, 0, stream>>>(qkv_w,  qkvw_b,   (size_t)NQKV * HIDDEN);
    k_f32_to_bf16<<<4096, 256, 0, stream>>>(o_w,    ow_b,     (size_t)HIDDEN * HQ * HDIM);

    // 2) QKV projection: (2048x2880) @ (5120x2880)^T -> fp32 qkv
    {
        dim3 grid(NQKV / GBN, S_LEN / GBM);      // 20 x 16
        k_gemm_xwt<<<grid, 256, 0, stream>>>(hidden_b, qkvw_b, qkv_f, S_LEN, NQKV, HIDDEN);
    }

    // 3) RoPE + split heads -> bf16 q/k/v
    k_rope_split<<<S_LEN, 128, 0, stream>>>(qkv_f, positions, q_b, k_b, v_b);

    // 4) sliding-window attention with sinks -> bf16 attn (S x HQ*D)
    {
        dim3 grid(HQ, S_LEN / 16);               // 64 x 128 blocks, 1 wave each
        k_attn<<<grid, 32, 0, stream>>>(q_b, k_b, v_b, sinks, attn_b);
    }

    // 5) O projection: (2048x4096) @ (2880x4096)^T -> fp32 out
    {
        dim3 grid((HIDDEN + GBN - 1) / GBN, S_LEN / GBM);  // 12 x 16
        k_gemm_xwt<<<grid, 256, 0, stream>>>(attn_b, ow_b, outp, S_LEN, HIDDEN, HQ * HDIM);
    }
}
